// Model_1786706395657
// MI455X (gfx1250) — matrix operand order repaired and checked by reading, not compiled, not run on hardware
//
#include <hip/hip_runtime.h>
#include <hip/hip_bf16.h>

typedef __attribute__((ext_vector_type(16))) _Float16 v16h;
typedef __attribute__((ext_vector_type(8)))  float    v8f;

#define BDIM 64
#define LDIM 512
#define NDIM 321
#define PDIM 720
#define EDIM 8
#define RDIM 32
#define CID  32
#define HDIM 64
#define MROWS (BDIM * NDIM)      // 20544
#define KHC   (EDIM * RDIM)      // 256 hidden cols
#define MBLK  64                 // rows per workgroup
#define NBLKS (MROWS / MBLK)     // 321 workgroups
#define ATILE (4 * 16 * 32 * 16) // halves of A fragments per block (64KB)

static __device__ __forceinline__ v8f wmma_f16(v16h a, v16h b, v8f c) {
    return __builtin_amdgcn_wmma_f32_16x16x32_f16(false, a, false, b, (short)0, c,
                                                  false, false);
}

// ---------------- mean / unbiased-std per (b, n) ----------------
__global__ void stats_kernel(const float* __restrict__ x,
                             float* __restrict__ mean, float* __restrict__ stdp) {
    int n = blockIdx.x * 64 + threadIdx.x;
    int b = blockIdx.y;
    if (n >= NDIM) return;
    const float* xb = x + (size_t)b * LDIM * NDIM + n;
    float s = 0.f, s2 = 0.f;
    for (int l = 0; l < LDIM; ++l) {
        float v = xb[(size_t)l * NDIM];
        s += v; s2 += v * v;
    }
    float m   = s * (1.0f / LDIM);
    float var = (s2 - (float)LDIM * m * m) * (1.0f / (LDIM - 1));
    var = fmaxf(var, 0.f);
    mean[b * NDIM + n] = m;
    stdp[b * NDIM + n] = sqrtf(var) + 1e-6f;
}

// ---------------- router MLP + softmax + gated bias ----------------
__global__ void router_kernel(const float* __restrict__ ident,
                              const float* __restrict__ rw1, const float* __restrict__ rb1,
                              const float* __restrict__ rw2, const float* __restrict__ rb2,
                              const float* __restrict__ bias,
                              float* __restrict__ gate, float* __restrict__ gb) {
    __shared__ float hid[HDIM];
    __shared__ float g[EDIM];
    int n = blockIdx.x, t = threadIdx.x;      // 64 threads
    float acc = rb1[t];
    for (int c = 0; c < CID; ++c) acc += ident[n * CID + c] * rw1[c * HDIM + t];
    hid[t] = fmaxf(acc, 0.f);
    __syncthreads();
    if (t < EDIM) {
        float lg = rb2[t];
        for (int h = 0; h < HDIM; ++h) lg += hid[h] * rw2[h * EDIM + t];
        g[t] = lg;
    }
    __syncthreads();
    if (t == 0) {
        float mx = g[0];
        for (int e = 1; e < EDIM; ++e) mx = fmaxf(mx, g[e]);
        float sum = 0.f;
        for (int e = 0; e < EDIM; ++e) { g[e] = expf(g[e] - mx); sum += g[e]; }
        float inv = 1.0f / sum;
        for (int e = 0; e < EDIM; ++e) { g[e] *= inv; gate[n * EDIM + e] = g[e]; }
    }
    __syncthreads();
    for (int p = t; p < PDIM; p += 64) {
        float a = 0.f;
        for (int e = 0; e < EDIM; ++e) a += g[e] * bias[e * PDIM + p];
        gb[n * PDIM + p] = a;
    }
}

// ---- normalize + transpose + f16 + A-fragment swizzle: xa[rt][kc][lane][16] ----
__global__ void prep_xn(const float* __restrict__ x, const float* __restrict__ mean,
                        const float* __restrict__ stdp, _Float16* __restrict__ xa) {
    __shared__ __align__(32) _Float16 tile[16 * LDIM];   // [kc(16)][lane(32)][i(16)]
    __shared__ float mS[16], rS[16];
    int rt = blockIdx.x, tid = threadIdx.x;
    if (tid < 16) {
        int flat = rt * 16 + tid;
        mS[tid] = mean[flat];
        rS[tid] = 1.0f / stdp[flat];
    }
    __syncthreads();
    for (int idx = tid; idx < 16 * LDIM; idx += 256) {
        int l = idx >> 4;            // k (time) index
        int r = idx & 15;            // row within tile -> consecutive n: coalesced
        int flat = rt * 16 + r;
        int b = flat / NDIM;
        int n = flat - b * NDIM;
        float v = x[((size_t)b * LDIM + l) * NDIM + n];
        v = (v - mS[r]) * rS[r];
        int kc = l >> 5, kk = l & 31;
        int lane = r + (((kk >> 3) & 1) << 4);
        int i = ((kk >> 4) << 3) + (kk & 7);
        tile[(kc * 32 + lane) * 16 + i] = (_Float16)v;
    }
    __syncthreads();
    float4* dst = (float4*)(xa + (size_t)rt * (16 * LDIM));
    const float4* src = (const float4*)tile;
    for (int idx = tid; idx < (16 * LDIM) / 8; idx += 256) dst[idx] = src[idx];
}

// ---- w1[E,L,R] -> B-fragments wb1[kc(16)][ht(16)][lane(32)][16] ----
__global__ void prep_w1(const float* __restrict__ w1, _Float16* __restrict__ wb1) {
    int idx = blockIdx.x * 256 + threadIdx.x;          // 131072 halves
    if (idx >= LDIM * KHC) return;
    int i    = idx & 15;
    int lane = (idx >> 4) & 31;
    int ht   = (idx >> 9) & 15;
    int kc   = idx >> 13;
    int hcol = ht * 16 + (lane & 15);
    int e = hcol >> 5, r = hcol & 31;
    int k = kc * 32 + ((i >> 3) << 4) + ((lane >> 4) << 3) + (i & 7);
    wb1[idx] = (_Float16)w1[((size_t)e * LDIM + k) * RDIM + r];
}

// ---- w2[E,R,P] -> transposed A-fragments wa2[pt(45)][kc(8)][lane(32)][16] ----
__global__ void prep_w2(const float* __restrict__ w2, _Float16* __restrict__ wa2) {
    int idx = blockIdx.x * 256 + threadIdx.x;          // 184320 halves
    if (idx >= KHC * PDIM) return;
    int i    = idx & 15;
    int lane = (idx >> 4) & 31;
    int kc   = (idx >> 9) & 7;
    int pt   = idx >> 12;
    int p = pt * 16 + (lane & 15);
    int koff = ((lane >> 4) << 4) + i;
    int k = kc * 32 + koff;
    int e = k >> 5, r = k & 31;
    wa2[idx] = (_Float16)w2[((size_t)e * RDIM + r) * PDIM + p];
}

// ---------------- fused two-stage WMMA GEMM ----------------
__global__ __launch_bounds__(256, 1)
void moe_gemm(const _Float16* __restrict__ xa, const _Float16* __restrict__ wb1,
              const _Float16* __restrict__ wa2, const float* __restrict__ gate,
              const float* __restrict__ gb, const float* __restrict__ mean,
              const float* __restrict__ stdp, float* __restrict__ out) {
    // A fragments for 64 rows x 512 k, async-copied from global (64KB, linear)
    __shared__ __align__(32) _Float16 xaS[ATILE];
    // hB: stage-2 B fragments [kc(8)][nt(4)][lane(32)][i(16)] = 32KB
    __shared__ __align__(32) _Float16 hB[8 * 4 * 32 * 16];
    __shared__ float gateS[MBLK * EDIM];
    __shared__ float meanS[MBLK], stdS[MBLK];

    int tid = threadIdx.x;
    int lane = tid & 31, w = tid >> 5;           // 8 waves of 32
    int rowbase = blockIdx.x * MBLK;

    // ---- async copy: this block's A fragments (one contiguous 64KB region) ----
    {
        const char* gsrc = (const char*)(xa + (size_t)blockIdx.x * ATILE);
        unsigned ldsbase = (unsigned)(uintptr_t)(void*)xaS;
        #pragma unroll
        for (int it = 0; it < 16; ++it) {
            unsigned off = (unsigned)(tid + it * 256) * 16u;
            unsigned lds_addr = ldsbase + off;
            unsigned long long ga = (unsigned long long)(uintptr_t)(gsrc + off);
            asm volatile("global_load_async_to_lds_b128 %0, %1, off"
                         :: "v"(lds_addr), "v"(ga) : "memory");
        }
    }

    for (int idx = tid; idx < MBLK * EDIM; idx += 256) {
        int r = idx >> 3, e = idx & 7;
        int flat = rowbase + r;
        int n = flat % NDIM;
        gateS[idx] = gate[n * EDIM + e];
    }
    if (tid < MBLK) {
        meanS[tid] = mean[rowbase + tid];
        stdS[tid]  = stdp[rowbase + tid];
    }
    asm volatile("s_wait_asynccnt 0x0" ::: "memory");
    __syncthreads();

    const v16h* AS    = (const v16h*)xaS;
    const v16h* Bbase = (const v16h*)wb1;
    const v8f   zero  = {0.f, 0.f, 0.f, 0.f, 0.f, 0.f, 0.f, 0.f};
    int jrow_off = (lane >> 4) << 3;             // +8 for hi half-wave
    int nloc = lane & 15;

    // ---- stage 1: h[64 x 256] = xn x w1^T ; wave w owns expert e = w ----
    v8f acc[4][2];
    #pragma unroll
    for (int rt = 0; rt < 4; ++rt) { acc[rt][0] = zero; acc[rt][1] = zero; }

    for (int kc = 0; kc < 16; ++kc) {
        v16h b0 = Bbase[(kc * 16 + 2 * w + 0) * 32 + lane];
        v16h b1 = Bbase[(kc * 16 + 2 * w + 1) * 32 + lane];
        #pragma unroll
        for (int rt = 0; rt < 4; ++rt) {
            v16h a = AS[((rt * 16 + kc) * 32) + lane];
            acc[rt][0] = wmma_f16(a, b0, acc[rt][0]);
            acc[rt][1] = wmma_f16(a, b1, acc[rt][1]);
        }
    }

    // gate-scale, f16-convert, scatter into stage-2 B-fragment layout
    #pragma unroll
    for (int rt = 0; rt < 4; ++rt) {
        #pragma unroll
        for (int ht = 0; ht < 2; ++ht) {
            int ksel = ht << 4;                  // (hcol & 31) >= 16 ? 16 : 0
            #pragma unroll
            for (int j = 0; j < 8; ++j) {
                int rl = rt * 16 + j + jrow_off; // local row 0..63
                float g = gateS[rl * EDIM + w];  // expert == wave id
                int dl = (rl & 15) + ksel;
                int nt = rl >> 4;
                hB[(((w * 4 + nt) * 32) + dl) * 16 + nloc] =
                    (_Float16)(acc[rt][ht][j] * g);
            }
        }
    }
    __syncthreads();

    // ---- stage 2: y^T[p, row] = w2^T x h ; output tile is p x n ----
    const v16h* A2 = (const v16h*)wa2;
    const v16h* HB = (const v16h*)hB;
    for (int pt = w; pt < PDIM / 16; pt += 8) {
        if (pt + 8 < PDIM / 16)
            __builtin_prefetch((const void*)&A2[((pt + 8) * 8) * 32 + lane], 0, 3);
        v8f c[4];
        #pragma unroll
        for (int nt = 0; nt < 4; ++nt) c[nt] = zero;
        for (int kc = 0; kc < 8; ++kc) {
            v16h a = A2[((pt * 8 + kc) * 32) + lane];
            #pragma unroll
            for (int nt = 0; nt < 4; ++nt) {
                v16h bb = HB[((kc * 4 + nt) * 32) + lane];
                c[nt] = wmma_f16(a, bb, c[nt]);
            }
        }
        #pragma unroll
        for (int nt = 0; nt < 4; ++nt) {
            int rl = nt * 16 + nloc;
            int flat = rowbase + rl;
            int b = flat / NDIM;
            int n = flat - b * NDIM;
            float sd = stdS[rl], mn = meanS[rl];
            #pragma unroll
            for (int j = 0; j < 8; ++j) {
                int p = pt * 16 + j + jrow_off;
                float y = c[nt][j] + gb[n * PDIM + p];
                out[((size_t)b * PDIM + p) * NDIM + n] = y * sd + mn;
            }
        }
    }
}

extern "C" void kernel_launch(void* const* d_in, const int* in_sizes, int n_in,
                              void* d_out, int out_size, void* d_ws, size_t ws_size,
                              hipStream_t stream) {
    const float* x     = (const float*)d_in[0];
    const float* ident = (const float*)d_in[1];
    const float* rw1   = (const float*)d_in[2];
    const float* rb1   = (const float*)d_in[3];
    const float* rw2   = (const float*)d_in[4];
    const float* rb2   = (const float*)d_in[5];
    const float* w1    = (const float*)d_in[6];
    const float* w2    = (const float*)d_in[7];
    const float* bias  = (const float*)d_in[8];
    float* out = (float*)d_out;

    char* ws = (char*)d_ws;
    size_t off = 0;
    auto carve = [&](size_t bytes) -> char* {
        char* p = ws + off;
        off += (bytes + 255) & ~(size_t)255;
        return p;
    };
    float*    meanP = (float*)carve((size_t)MROWS * 4);
    float*    stdP  = (float*)carve((size_t)MROWS * 4);
    float*    gateP = (float*)carve((size_t)NDIM * EDIM * 4);
    float*    gbP   = (float*)carve((size_t)NDIM * PDIM * 4);
    _Float16* xaP   = (_Float16*)carve((size_t)MROWS * LDIM * 2);
    _Float16* wb1P  = (_Float16*)carve((size_t)LDIM * KHC * 2);
    _Float16* wa2P  = (_Float16*)carve((size_t)KHC * PDIM * 2);

    stats_kernel<<<dim3((NDIM + 63) / 64, BDIM), 64, 0, stream>>>(x, meanP, stdP);
    router_kernel<<<NDIM, 64, 0, stream>>>(ident, rw1, rb1, rw2, rb2, bias, gateP, gbP);
    prep_xn<<<MROWS / 16, 256, 0, stream>>>(x, meanP, stdP, xaP);
    prep_w1<<<(LDIM * KHC + 255) / 256, 256, 0, stream>>>(w1, wb1P);
    prep_w2<<<(KHC * PDIM + 255) / 256, 256, 0, stream>>>(w2, wa2P);
    moe_gemm<<<NBLKS, 256, 0, stream>>>(xaP, wb1P, wa2P, gateP, gbP, meanP, stdP, out);
}
